// BaseMHA_50268297232575
// MI455X (gfx1250) — matrix operand order repaired and checked by reading, not compiled, not run on hardware
//
#include <hip/hip_runtime.h>
#include <hip/hip_bf16.h>

// ---------------------------------------------------------------------------
// Causal multi-head attention (B=4, N=2048, D=1024, H=16, DH=64) for gfx1250.
// f16 WMMA (v_wmma_f32_16x16x32_f16) with f32 accumulation everywhere.
// Global->LDS staging uses gfx1250 async-to-LDS copies (ASYNCcnt) with LDS
// double buffering in the GEMMs; falls back to plain copies if the builtins
// are unavailable.
// ---------------------------------------------------------------------------

typedef __attribute__((ext_vector_type(16))) _Float16 v16h;
typedef __attribute__((ext_vector_type(8)))  _Float16 v8h;
typedef __attribute__((ext_vector_type(8)))  float    v8f;

#define BATCH   4
#define SEQ     2048
#define DIM     1024
#define HEADS   16
#define DHEAD   64
#define MROWS   (BATCH * SEQ)      // 8192
#define QKN     (3 * DIM)          // 3072
#define SCALE   0.125f             // DHEAD^-0.5

#ifndef __has_builtin
#define __has_builtin(x) 0
#endif

#if __has_builtin(__builtin_amdgcn_global_load_async_to_lds_b128)
#define HAVE_ASYNC_LDS 1
#else
#define HAVE_ASYNC_LDS 0
#endif
#if __has_builtin(__builtin_amdgcn_s_wait_asynccnt)
#define HAVE_WAIT_ASYNC 1
#else
#define HAVE_WAIT_ASYNC 0
#endif

#if HAVE_ASYNC_LDS
// Builtin prototype (per hipcc diagnostic): first arg is a pointer to a
// GCC-vector int4 in the global address space, second in LDS address space.
typedef int v4i_gcc __attribute__((__vector_size__(16)));
typedef __attribute__((address_space(1))) v4i_gcc gbl_v4i_t;
typedef __attribute__((address_space(3))) v4i_gcc lds_v4i_t;
#endif

// Copy 16 bytes global -> LDS. Async (ASYNCcnt-tracked, no VGPR staging) when
// available; plain vector copy otherwise.
__device__ inline void copy_b128_to_lds(const void* g, void* l) {
#if HAVE_ASYNC_LDS
    gbl_v4i_t* gp = (gbl_v4i_t*)(unsigned long long)g;
    lds_v4i_t* lp = (lds_v4i_t*)(unsigned)(unsigned long long)l;
    __builtin_amdgcn_global_load_async_to_lds_b128(gp, lp, 0, 0);
#else
    *(float4*)l = *(const float4*)g;
#endif
}

// Wait for this wave's outstanding async-to-LDS copies.
__device__ inline void wait_async0() {
#if HAVE_ASYNC_LDS
#if HAVE_WAIT_ASYNC
    __builtin_amdgcn_s_wait_asynccnt(0);
#else
    asm volatile("s_wait_asynccnt 0x0" ::: "memory");
#endif
#endif
}

union H16u { v16h v; v8h h[2]; };

// A-fragment (16x32 f16), row-major source with leading dim `ld` (halfs).
// Layout (ISA 7.12.2): lanes 0-15 row r hold K 0..7 / 16..23; lanes 16-31 row
// r hold K 8..15 / 24..31  ->  koff = (lane>>4)*8, second chunk at +16.
__device__ inline v16h load_frag_A(const _Float16* base, int ld, int lane) {
    int r    = lane & 15;
    int koff = (lane >> 4) * 8;
    const _Float16* p = base + r * ld + koff;
    H16u u;
    u.h[0] = *(const v8h*)(p);
    u.h[1] = *(const v8h*)(p + 16);
    return u.v;
}

// B-fragment (32x16 f16) from an N-major (transposed) tile BT[n][k], leading
// dim `ld`: lanes 0-15 col c hold K 0..7 / 16..23, lanes 16-31 col c hold K 8..15 / 24..31.
__device__ inline v16h load_frag_B(const _Float16* baseT, int ld, int lane) {
    int c  = lane & 15;
    int kb = (lane >> 4) * 8;
    const _Float16* p = baseT + c * ld + kb;
    H16u u;
    u.h[0] = *(const v8h*)(p);
    u.h[1] = *(const v8h*)(p + 16);
    return u.v;
}

__device__ inline v8f vzero8f() {
    v8f z = {0.f, 0.f, 0.f, 0.f, 0.f, 0.f, 0.f, 0.f};
    return z;
}

// ---------------------------------------------------------------------------
// Casting kernels (one-time, tiny vs. GEMM cost)
// ---------------------------------------------------------------------------
__global__ void cast_f32_to_f16(const float* __restrict__ s,
                                _Float16* __restrict__ d, int n) {
    int i = blockIdx.x * 256 + threadIdx.x;
    if (i < n) d[i] = (_Float16)s[i];
}

// s[K][N] (f32) -> d[N][K] (f16) so GEMM B-tiles are K-contiguous per column.
__global__ void transpose_cast_f16(const float* __restrict__ s,
                                   _Float16* __restrict__ d, int K, int N) {
    long i = (long)blockIdx.x * 256 + threadIdx.x;
    long total = (long)K * N;
    if (i < total) {
        long n = i / K;
        long k = i % K;
        d[i] = (_Float16)s[k * (long)N + n];
    }
}

// ---------------------------------------------------------------------------
// Tiled WMMA GEMM:  C[M,N] = A[M,K] (f16, row-major) @ BT[N,K]^T (f16)
// Block tile 128x128, 8 waves (4 M-strips x 2 N-strips), wave tile 32x64.
// Double-buffered LDS; next K-tile staged with async-to-LDS while current
// tile feeds the WMMAs. One barrier per K-step:
//   stage(i+1) is issued after the barrier of iteration i, and the last
//   readers of that buffer (iteration i-1) finished before that barrier.
// MODE 0: scatter into q/k/v f16 buffers [b,h,n,64] (Q scaled by 1/sqrt(dh)).
// MODE 1: write f32 into outF[M,N].
// ---------------------------------------------------------------------------
template <int MODE>
__global__ __launch_bounds__(256) void gemm_wmma_f16(
    const _Float16* __restrict__ A, const _Float16* __restrict__ BT,
    int M, int N, int K,
    _Float16* __restrict__ qv, _Float16* __restrict__ kv,
    _Float16* __restrict__ vv, float* __restrict__ outF) {
    __shared__ _Float16 As[2][128][40];  // +8 pad keeps rows 16B aligned
    __shared__ _Float16 Bs[2][128][40];

    const int tid  = threadIdx.x;
    const int lane = tid & 31;
    const int w    = tid >> 5;
    const int wm   = w & 3;            // 0..3  -> 32-row strip
    const int wn   = w >> 2;           // 0..1  -> 64-col strip
    const long bm  = (long)blockIdx.y * 128;
    const long bn  = (long)blockIdx.x * 128;

    v8f acc[2][4];
#pragma unroll
    for (int mi = 0; mi < 2; ++mi)
#pragma unroll
        for (int ni = 0; ni < 4; ++ni) acc[mi][ni] = vzero8f();

    const int r  = tid >> 1;           // 0..127 staging row
    const int co = (tid & 1) * 16;     // 0 / 16 halfs within 32-K chunk
    const _Float16* ga = A  + (bm + r) * (long)K + co;
    const _Float16* gb = BT + (bn + r) * (long)K + co;

    // Prologue: stage first K-tile into buffer 0.
    copy_b128_to_lds(ga, &As[0][r][co]);
    copy_b128_to_lds(ga + 8, &As[0][r][co + 8]);
    copy_b128_to_lds(gb, &Bs[0][r][co]);
    copy_b128_to_lds(gb + 8, &Bs[0][r][co + 8]);

    int ib = 0;
    for (int kt = 0; kt < K; kt += 32, ib ^= 1) {
        wait_async0();                 // this wave's staged tile is in LDS
        __syncthreads();               // everyone's tile is in LDS

        if (kt + 32 < K) {             // stage next tile into other buffer
            const _Float16* gan = ga + kt + 32;
            const _Float16* gbn = gb + kt + 32;
            copy_b128_to_lds(gan, &As[ib ^ 1][r][co]);
            copy_b128_to_lds(gan + 8, &As[ib ^ 1][r][co + 8]);
            copy_b128_to_lds(gbn, &Bs[ib ^ 1][r][co]);
            copy_b128_to_lds(gbn + 8, &Bs[ib ^ 1][r][co + 8]);
        }

        v16h af[2], bf[4];
#pragma unroll
        for (int mi = 0; mi < 2; ++mi)
            af[mi] = load_frag_A(&As[ib][wm * 32 + mi * 16][0], 40, lane);
#pragma unroll
        for (int ni = 0; ni < 4; ++ni)
            bf[ni] = load_frag_B(&Bs[ib][wn * 64 + ni * 16][0], 40, lane);
#pragma unroll
        for (int mi = 0; mi < 2; ++mi)
#pragma unroll
            for (int ni = 0; ni < 4; ++ni)
                acc[mi][ni] = __builtin_amdgcn_wmma_f32_16x16x32_f16(
                    false, af[mi], false, bf[ni], (short)0, acc[mi][ni],
                    false, false);
    }

    // Epilogue. C layout: VGPR e -> row e (lanes 0-15) or e+8 (lanes 16-31),
    // col = lane&15 within each 16x16 fragment.
    const int c  = lane & 15;
    const int rh = (lane >> 4) * 8;
#pragma unroll
    for (int mi = 0; mi < 2; ++mi) {
#pragma unroll
        for (int ni = 0; ni < 4; ++ni) {
#pragma unroll
            for (int e = 0; e < 8; ++e) {
                long row = bm + wm * 32 + mi * 16 + rh + e;
                long col = bn + wn * 64 + ni * 16 + c;
                float val = acc[mi][ni][e];
                if (MODE == 0) {
                    int which = (int)(col >> 10);       // 0=q 1=k 2=v
                    int d     = (int)(col & 1023);
                    int h     = d >> 6;
                    int dh    = d & 63;
                    long b    = row >> 11;              // /SEQ
                    long n    = row & 2047;
                    long o    = ((b * HEADS + h) * (long)SEQ + n) * DHEAD + dh;
                    if (which == 0)      qv[o] = (_Float16)(val * SCALE);
                    else if (which == 1) kv[o] = (_Float16)val;
                    else                 vv[o] = (_Float16)val;
                } else {
                    outF[row * (long)N + col] = val;
                }
            }
        }
    }
}

// ---------------------------------------------------------------------------
// Flash attention: one block = 64-query tile of one (b,h). 4 waves, 16 rows
// each. Stream 64-key tiles (causal: only blocks j <= qb), online softmax.
// K tile staged with async-to-LDS; V tile transposed manually (needs
// element scatter). Q pre-scaled by 1/sqrt(dh) at QKV epilogue.
// ---------------------------------------------------------------------------
#define ALD 72   // K/V LDS leading dim (halfs); 144B rows, 16B aligned

__global__ __launch_bounds__(128) void attn_flash_wmma(
    const _Float16* __restrict__ qh, const _Float16* __restrict__ kh,
    const _Float16* __restrict__ vh, _Float16* __restrict__ attn) {
    __shared__ _Float16 Ks[64][ALD];      // [key][dim]
    __shared__ _Float16 Vt[64][ALD];      // [dim][key]  (transposed V)
    __shared__ _Float16 Pl[4][16][64];    // per-wave probabilities

    const int tid  = threadIdx.x;
    const int lane = tid & 31;
    const int w    = tid >> 5;            // wave -> 16-row strip
    const int qb   = blockIdx.x;          // query block 0..31
    const int bh   = blockIdx.y;          // b*HEADS + h
    const long base = (long)bh * SEQ * DHEAD;
    const int qrow0 = qb * 64;

    // Q fragments for this wave's 16 rows (dims 0..31 and 32..63)
    v16h aQ[2];
    {
        const _Float16* qbase = qh + base + (long)(qrow0 + w * 16) * DHEAD;
        aQ[0] = load_frag_A(qbase,      DHEAD, lane);
        aQ[1] = load_frag_A(qbase + 32, DHEAD, lane);
    }

    v8f accO[4];
#pragma unroll
    for (int ni = 0; ni < 4; ++ni) accO[ni] = vzero8f();
    float mrow[8], lrow[8];
#pragma unroll
    for (int e = 0; e < 8; ++e) { mrow[e] = -3.0e38f; lrow[e] = 0.f; }

    const int c  = lane & 15;
    const int rh = (lane >> 4) * 8;

    for (int j = 0; j <= qb; ++j) {
        // ---- stage K tile (async) and transposed V tile -----------------
        {
            const int rr = tid >> 1;
            const int cc = (tid & 1) * 32;
            const _Float16* gk =
                kh + base + (long)(j * 64 + rr) * DHEAD + cc;
            copy_b128_to_lds(gk,      &Ks[rr][cc + 0]);
            copy_b128_to_lds(gk + 8,  &Ks[rr][cc + 8]);
            copy_b128_to_lds(gk + 16, &Ks[rr][cc + 16]);
            copy_b128_to_lds(gk + 24, &Ks[rr][cc + 24]);
            union { float4 f4[4]; _Float16 hh[32]; } uv;
            const float4* gv =
                (const float4*)(vh + base + (long)(j * 64 + rr) * DHEAD + cc);
            uv.f4[0] = gv[0]; uv.f4[1] = gv[1];
            uv.f4[2] = gv[2]; uv.f4[3] = gv[3];
#pragma unroll
            for (int i = 0; i < 32; ++i) Vt[cc + i][rr] = uv.hh[i];
        }
        wait_async0();
        __syncthreads();

        // ---- S = Q @ K^T (scores already scaled via Q) ------------------
        v8f accS[4];
#pragma unroll
        for (int ni = 0; ni < 4; ++ni) accS[ni] = vzero8f();
#pragma unroll
        for (int kc = 0; kc < 2; ++kc) {
#pragma unroll
            for (int ni = 0; ni < 4; ++ni) {
                v16h bK = load_frag_B(&Ks[ni * 16][0] + kc * 32, ALD, lane);
                accS[ni] = __builtin_amdgcn_wmma_f32_16x16x32_f16(
                    false, aQ[kc], false, bK, (short)0, accS[ni], false, false);
            }
        }

        // ---- causal mask on the diagonal block --------------------------
        if (j == qb) {
#pragma unroll
            for (int ni = 0; ni < 4; ++ni) {
                int col = ni * 16 + c;
#pragma unroll
                for (int e = 0; e < 8; ++e) {
                    int row = w * 16 + rh + e;
                    if (col > row) accS[ni][e] = -1.0e30f;
                }
            }
        }

        // ---- online softmax update --------------------------------------
        float mnew[8], rs[8], psum[8];
#pragma unroll
        for (int e = 0; e < 8; ++e) {
            float rm = accS[0][e];
#pragma unroll
            for (int ni = 1; ni < 4; ++ni) rm = fmaxf(rm, accS[ni][e]);
            rm = fmaxf(rm, __shfl_xor(rm, 1, 32));
            rm = fmaxf(rm, __shfl_xor(rm, 2, 32));
            rm = fmaxf(rm, __shfl_xor(rm, 4, 32));
            rm = fmaxf(rm, __shfl_xor(rm, 8, 32));
            mnew[e] = fmaxf(mrow[e], rm);
            rs[e]   = __expf(mrow[e] - mnew[e]);
            psum[e] = 0.f;
        }
#pragma unroll
        for (int ni = 0; ni < 4; ++ni) {
#pragma unroll
            for (int e = 0; e < 8; ++e) {
                float p = __expf(accS[ni][e] - mnew[e]);
                psum[e] += p;
                Pl[w][rh + e][ni * 16 + c] = (_Float16)p;
            }
        }
#pragma unroll
        for (int e = 0; e < 8; ++e) {
            float s = psum[e];
            s += __shfl_xor(s, 1, 32);
            s += __shfl_xor(s, 2, 32);
            s += __shfl_xor(s, 4, 32);
            s += __shfl_xor(s, 8, 32);
            lrow[e] = lrow[e] * rs[e] + s;
            mrow[e] = mnew[e];
        }
#pragma unroll
        for (int ni = 0; ni < 4; ++ni)
#pragma unroll
            for (int e = 0; e < 8; ++e) accO[ni][e] *= rs[e];
        __syncthreads();   // P (and Vt) visible before P@V

        // ---- O += P @ V -------------------------------------------------
#pragma unroll
        for (int kc = 0; kc < 2; ++kc) {
            v16h aP = load_frag_A(&Pl[w][0][0] + kc * 32, 64, lane);
#pragma unroll
            for (int ni = 0; ni < 4; ++ni) {
                v16h bV = load_frag_B(&Vt[ni * 16][0] + kc * 32, ALD, lane);
                accO[ni] = __builtin_amdgcn_wmma_f32_16x16x32_f16(
                    false, aP, false, bV, (short)0, accO[ni], false, false);
            }
        }
        __syncthreads();   // before next iteration overwrites Ks/Vt
    }

    // ---- normalize + write [b, n, h*64+dh] as f16 -----------------------
    const int b = bh >> 4;
    const int h = bh & 15;
#pragma unroll
    for (int ni = 0; ni < 4; ++ni) {
#pragma unroll
        for (int e = 0; e < 8; ++e) {
            int  qr  = qrow0 + w * 16 + rh + e;
            int  dh  = ni * 16 + c;
            long o   = ((long)(b * SEQ + qr)) * DIM + h * DHEAD + dh;
            attn[o]  = (_Float16)(accO[ni][e] / lrow[e]);
        }
    }
}

// ---------------------------------------------------------------------------
// Launch
// ---------------------------------------------------------------------------
extern "C" void kernel_launch(void* const* d_in, const int* in_sizes, int n_in,
                              void* d_out, int out_size, void* d_ws,
                              size_t ws_size, hipStream_t stream) {
    const float* x    = (const float*)d_in[0];   // [4,2048,1024]
    const float* wqkv = (const float*)d_in[1];   // [1024,3072]
    const float* wout = (const float*)d_in[2];   // [1024,1024]
    float* out = (float*)d_out;                  // [4,2048,1024]

    char* ws = (char*)d_ws;
    size_t off = 0;
    _Float16* xh     = (_Float16*)(ws + off); off += (size_t)MROWS * DIM * 2;
    _Float16* wqkvT  = (_Float16*)(ws + off); off += (size_t)QKN * DIM * 2;
    _Float16* woutT  = (_Float16*)(ws + off); off += (size_t)DIM * DIM * 2;
    _Float16* qh     = (_Float16*)(ws + off); off += (size_t)MROWS * DIM * 2;
    _Float16* kh     = (_Float16*)(ws + off); off += (size_t)MROWS * DIM * 2;
    _Float16* vh     = (_Float16*)(ws + off); off += (size_t)MROWS * DIM * 2;
    _Float16* attnB  = (_Float16*)(ws + off); off += (size_t)MROWS * DIM * 2;
    (void)ws_size; (void)in_sizes; (void)n_in; (void)out_size;

    // 1) casts
    {
        int n = MROWS * DIM;
        cast_f32_to_f16<<<(n + 255) / 256, 256, 0, stream>>>(x, xh, n);
    }
    {
        int n = DIM * QKN;
        transpose_cast_f16<<<(n + 255) / 256, 256, 0, stream>>>(wqkv, wqkvT,
                                                                DIM, QKN);
    }
    {
        int n = DIM * DIM;
        transpose_cast_f16<<<(n + 255) / 256, 256, 0, stream>>>(wout, woutT,
                                                                DIM, DIM);
    }

    // 2) QKV projection + head scatter (Q scaled)
    gemm_wmma_f16<0><<<dim3(QKN / 128, MROWS / 128), 256, 0, stream>>>(
        xh, wqkvT, MROWS, QKN, DIM, qh, kh, vh, nullptr);

    // 3) causal flash attention
    attn_flash_wmma<<<dim3(SEQ / 64, BATCH * HEADS), 128, 0, stream>>>(
        qh, kh, vh, attnB);

    // 4) output projection -> f32
    gemm_wmma_f16<1><<<dim3(DIM / 128, MROWS / 128), 256, 0, stream>>>(
        attnB, woutT, MROWS, DIM, DIM, nullptr, nullptr, nullptr, out);
}
